// WeightConv_35433480192486
// MI455X (gfx1250) — matrix operand order repaired and checked by reading, not compiled, not run on hardware
//
#include <hip/hip_runtime.h>
#include <hip/hip_bf16.h>

typedef __attribute__((ext_vector_type(16))) __bf16 v16bf;
typedef __attribute__((ext_vector_type(8)))  __bf16 v8bf;
typedef __attribute__((ext_vector_type(8)))  float  v8f;

static constexpr int kB    = 32;    // batch
static constexpr int kINP  = 256;
static constexpr int kOUP  = 256;
static constexpr int kHW   = 784;   // 28*28
static constexpr int kMOUP = 512;   // M*OUP
static constexpr int kIGAP = 16;

// Workspace layout (bytes):
//   [0, 65536)                      h        : 32*512 f32
//   [65536, 65536+1179648)          Wt       : 9*256*256 bf16
//   [+16384 guard]                  front guard for x_t (conv shifts up to -29 px)
//   [1261568, +12845056)            x_t      : [32][784][256] bf16 (pixel-major)
//   [+16384 guard]                  back guard
static constexpr size_t kOffH   = 0;
static constexpr size_t kOffWt  = 65536;
static constexpr size_t kOffXt  = 65536 + 1179648 + 16384;   // data start (guarded)

// ---------------------------------------------------------------------------
// Kernel 1: h[b, j] = sigmoid( x_gap[b,:] . w1[j,:] + b1[j] )   -> [32, 512]
// ---------------------------------------------------------------------------
__global__ void wc_h_kernel(const float* __restrict__ xg,
                            const float* __restrict__ w1,
                            const float* __restrict__ b1,
                            float* __restrict__ h) {
  int idx = blockIdx.x * blockDim.x + threadIdx.x;
  if (idx >= kB * kMOUP) return;
  int b = idx >> 9;
  int j = idx & 511;
  const float* g = xg + b * kIGAP;
  const float* w = w1 + j * kIGAP;
  float s = b1[j];
#pragma unroll
  for (int i = 0; i < kIGAP; ++i) s = fmaf(g[i], w[i], s);
  h[idx] = 1.0f / (1.0f + __expf(-s));
}

// ---------------------------------------------------------------------------
// Kernel 2: Wt[tap][o][c] = bf16( w2[(o*256 + c)*9 + tap] )
// ---------------------------------------------------------------------------
__global__ void wc_wt_kernel(const float* __restrict__ w2,
                             unsigned short* __restrict__ wt_raw) {
  int idx = blockIdx.x * blockDim.x + threadIdx.x;   // [0, 9*65536)
  int kp = idx >> 16;
  int oc = idx & 65535;
  int o  = oc >> 8;
  int c  = oc & 255;
  __bf16 v = (__bf16)w2[(o * 256 + c) * 9 + kp];
  wt_raw[idx] = __builtin_bit_cast(unsigned short, v);
}

// ---------------------------------------------------------------------------
// Kernel 3: x_t[b][s][c] = bf16(x[b][c][s]) — LDS tile transpose (64c x 16s).
// ---------------------------------------------------------------------------
__global__ void __launch_bounds__(256)
wc_xt_kernel(const float* __restrict__ x, unsigned short* __restrict__ xt) {
  __shared__ float tile[64][17];
  const int blk  = blockIdx.x;       // 32 * 4 * 49 blocks
  const int st   = blk % 49;
  const int rest = blk / 49;
  const int ct   = rest & 3;
  const int b    = rest >> 2;
  const int t    = threadIdx.x;
  const int sbase = st * 16, cbase = ct * 64;

  const int cr = t >> 4;             // 0..15
  const int sc = t & 15;             // 0..15
  const float* xp = x + ((size_t)(b * kINP + cbase + cr)) * kHW + sbase + sc;
#pragma unroll
  for (int it = 0; it < 4; ++it)
    tile[cr + 16 * it][sc] = xp[(size_t)(16 * it) * kHW];
  __syncthreads();

  const int cc2 = t & 63;            // 0..63
  const int sr  = t >> 6;            // 0..3
  unsigned short* op = xt + ((size_t)(b * kHW + sbase + sr)) * kINP + cbase + cc2;
#pragma unroll
  for (int it = 0; it < 4; ++it) {
    __bf16 v = (__bf16)tile[cc2][sr + 4 * it];
    op[(size_t)(4 * it) * kINP] = __builtin_bit_cast(unsigned short, v);
  }
}

// ---------------------------------------------------------------------------
// Fragment load helpers (16B-aligned b128 pairs, base + immediate offsets)
// ---------------------------------------------------------------------------
__device__ __forceinline__ v16bf load_a_frag(const __bf16* p) {
  union { struct { v8bf lo; v8bf hi; } s; v16bf v; } u;
  u.s.lo = *(const v8bf*)(p);        // K = +0..7
  u.s.hi = *(const v8bf*)(p + 8);    // K = +8..15
  return u.v;
}
__device__ __forceinline__ v16bf load_b_frag(const __bf16* p) {
  union { struct { v8bf lo; v8bf hi; } s; v16bf v; } u;
  u.s.lo = *(const v8bf*)(p);        // K = +0..7
  u.s.hi = *(const v8bf*)(p + 8);    // K = +8..15 (contiguous 32B)
  return u.v;
}

// ---------------------------------------------------------------------------
// Kernel 4: implicit-GEMM conv; 288 WMMAs per wave, software-pipelined B and
// grouped A loads. B loads are UNCONDITIONAL (guard pages make any border
// address valid) followed by a vector select -> plain loads + v_cndmask,
// no exec-mask manipulation in the hot loop.
// ---------------------------------------------------------------------------
__global__ void __launch_bounds__(256)
wc_conv_kernel(const unsigned short* __restrict__ xt_raw,
               const unsigned short* __restrict__ wt_raw,
               const float* __restrict__ h,
               float* __restrict__ out) {
  const __bf16* xt = (const __bf16*)xt_raw;
  const __bf16* wt = (const __bf16*)wt_raw;

  const int lane = threadIdx.x & 31;
  const int wave = threadIdx.x >> 5;
  const int tile = blockIdx.x * 8 + wave;   // < 32 * 4 * 49 = 6272
  const int n_t  = tile % 49;
  const int rest = tile / 49;
  const int mblk = rest & 3;                // 4 M-blocks of 64 out-channels
  const int b    = rest >> 2;               // sample

  const int lrow = lane & 15;
  const int hi16 = lane >> 4;               // 0 | 1
  const int s    = n_t * 16 + lrow;         // pixel (B/C column)
  const int sy   = s / 28;
  const int sx   = s - sy * 28;

  // Lane-invariant bases; all tap/chunk offsets fold into immediates.
  const __bf16* btp = xt + (size_t)(b * kHW + s) * kINP + hi16 * 16;      // B base
  const __bf16* atp = wt + (size_t)(mblk * 64 + lrow) * kINP + hi16 * 16; // A base

  // Per-tap border masks (compile-time indexed after unrolling).
  bool inb[9];
#pragma unroll
  for (int kp = 0; kp < 9; ++kp) {
    const int dy = kp / 3 - 1, dx = kp % 3 - 1;
    inb[kp] = (((unsigned)(sy + dy) < 28u) & ((unsigned)(sx + dx) < 28u));
  }

  v8f acc0[4], acc1[4];
  const v8f vz = {0.f, 0.f, 0.f, 0.f, 0.f, 0.f, 0.f, 0.f};
#pragma unroll
  for (int j = 0; j < 4; ++j) { acc0[j] = vz; acc1[j] = vz; }

  v16bf zbf;
#pragma unroll
  for (int e = 0; e < 16; ++e) zbf[e] = (__bf16)0.0f;

  // Software pipeline over 72 (tap, chunk) steps: load B(i+1) before WMMA(i).
  auto b_off = [](int kp, int cc) {
    const int dy = kp / 3 - 1, dx = kp % 3 - 1;
    return (dy * 28 + dx) * kINP + cc * 32;
  };

  // Prologue: unconditional load (guard-safe) + cndmask select.
  v16bf braw = load_b_frag(btp + b_off(0, 0));
  v16bf bcur = inb[0] ? braw : zbf;

#pragma unroll
  for (int idx = 0; idx < 72; ++idx) {
    const int kp = idx / 8;
    const int cc = idx & 7;
    const int c0 = cc * 32;

    // Issue all four A-fragment loads for this chunk (one clause).
    const __bf16* ap = atp + kp * 65536 + c0;
    const v16bf a0 = load_a_frag(ap);
    const v16bf a1 = load_a_frag(ap + 4096);
    const v16bf a2 = load_a_frag(ap + 8192);
    const v16bf a3 = load_a_frag(ap + 12288);

    // Prefetch next chunk's B fragment: unconditional load, then select.
    v16bf bnext = zbf;
    if (idx < 71) {
      const int kpn = (idx + 1) / 8;
      const int ccn = (idx + 1) & 7;
      const v16bf bload = load_b_frag(btp + b_off(kpn, ccn));  // always executed
      bnext = inb[kpn] ? bload : zbf;                          // v_cndmask, no branch
    }

    if (cc < 4) {
      acc0[0] = __builtin_amdgcn_wmma_f32_16x16x32_bf16(false, a0, false, bcur, (short)0, acc0[0], false, false);
      acc0[1] = __builtin_amdgcn_wmma_f32_16x16x32_bf16(false, a1, false, bcur, (short)0, acc0[1], false, false);
      acc0[2] = __builtin_amdgcn_wmma_f32_16x16x32_bf16(false, a2, false, bcur, (short)0, acc0[2], false, false);
      acc0[3] = __builtin_amdgcn_wmma_f32_16x16x32_bf16(false, a3, false, bcur, (short)0, acc0[3], false, false);
    } else {
      acc1[0] = __builtin_amdgcn_wmma_f32_16x16x32_bf16(false, a0, false, bcur, (short)0, acc1[0], false, false);
      acc1[1] = __builtin_amdgcn_wmma_f32_16x16x32_bf16(false, a1, false, bcur, (short)0, acc1[1], false, false);
      acc1[2] = __builtin_amdgcn_wmma_f32_16x16x32_bf16(false, a2, false, bcur, (short)0, acc1[2], false, false);
      acc1[3] = __builtin_amdgcn_wmma_f32_16x16x32_bf16(false, a3, false, bcur, (short)0, acc1[3], false, false);
    }
    bcur = bnext;
  }

  // Epilogue: y[b,o,s] = h[b,2o]*acc0 + h[b,2o+1]*acc1
  // C/D layout: VGPR r -> M = r + 8*hi16, N = lane&15.
  const float* hb = h + b * kMOUP;
  float* ob = out + (size_t)b * kOUP * kHW + s;
#pragma unroll
  for (int j = 0; j < 4; ++j) {
#pragma unroll
    for (int r = 0; r < 8; ++r) {
      const int o = mblk * 64 + j * 16 + r + hi16 * 8;
      const float2 hv = *(const float2*)(hb + 2 * o);
      ob[(size_t)o * kHW] = hv.x * acc0[j][r] + hv.y * acc1[j][r];
    }
  }
}

// ---------------------------------------------------------------------------
extern "C" void kernel_launch(void* const* d_in, const int* in_sizes, int n_in,
                              void* d_out, int out_size, void* d_ws, size_t ws_size,
                              hipStream_t stream) {
  const float* x  = (const float*)d_in[0];   // [32,256,28,28]
  const float* xg = (const float*)d_in[1];   // [32,16,1,1]
  const float* w1 = (const float*)d_in[2];   // [512,16]
  const float* b1 = (const float*)d_in[3];   // [512]
  const float* w2 = (const float*)d_in[4];   // [589824]
  float* out = (float*)d_out;                // [32,256,28,28]

  float*          hbuf = (float*)((char*)d_ws + kOffH);
  unsigned short* wtb  = (unsigned short*)((char*)d_ws + kOffWt);
  unsigned short* xtb  = (unsigned short*)((char*)d_ws + kOffXt);

  wc_h_kernel<<<(kB * kMOUP + 255) / 256, 256, 0, stream>>>(xg, w1, b1, hbuf);
  wc_wt_kernel<<<(9 * 65536) / 256, 256, 0, stream>>>(w2, wtb);
  wc_xt_kernel<<<kB * 4 * 49, 256, 0, stream>>>(x, xtb);
  wc_conv_kernel<<<784, 256, 0, stream>>>(xtb, wtb, hbuf, out);
}
